// AugGraphConv_49692771615017
// MI455X (gfx1250) — matrix operand order repaired and checked by reading, not compiled, not run on hardware
//
#include <hip/hip_runtime.h>
#include <hip/hip_bf16.h>
#include <math.h>

// ---------------------------------------------------------------------------
// AugGraphConv for MI455X (gfx1250, wave32, WMMA).
// GEMMs: f16 in / f32 acc via v_wmma_f32_16x16x32_f16 (B frags preloaded per
// K-step so the 8 WMMAs issue back-to-back after one load clause).
// Edge phase: f32 + atomics (bandwidth bound, L2-resident per relation).
// ---------------------------------------------------------------------------

typedef __attribute__((ext_vector_type(16))) _Float16 v16h;
typedef __attribute__((ext_vector_type(8)))  _Float16 v8h;
typedef __attribute__((ext_vector_type(8)))  float    v8f;

#define D_   128
#define H_   8
#define C_   16
#define R_   5
#define NEGV (-1e9f)

__device__ __forceinline__ unsigned fkey(float f) {
  unsigned u = __float_as_uint(f);
  return (u & 0x80000000u) ? ~u : (u | 0x80000000u);
}
__device__ __forceinline__ float funkey(unsigned k) {
  return (k & 0x80000000u) ? __uint_as_float(k ^ 0x80000000u)
                           : __uint_as_float(~k);
}
__device__ __forceinline__ float geluf(float x) {
  return 0.5f * x * (1.0f + erff(x * 0.70710678118654752f));
}
__device__ __forceinline__ float lreluf(float x) { return x > 0.0f ? x : 0.2f * x; }

// ---------------------------------------------------------------------------
// Pack the 7 [128,128] f32 weight matrices into per-lane WMMA B-fragment
// layout (f16). Fragment for (kstep ks, ntile nt, lane l):
//   col = nt*16 + (l&15); k = ks*32 + (l<16?0:16) + j, j=0..15 contiguous.
// One thread builds one 16-half fragment slice (7*4*8*32 = 7168 threads).
// ---------------------------------------------------------------------------
__global__ void pack_w_kernel(const float* __restrict__ Wself,
                              const float* __restrict__ Wword,
                              const float* __restrict__ Wcross,
                              _Float16* __restrict__ wf) {
  int t = blockIdx.x * blockDim.x + threadIdx.x;
  if (t >= 7 * 4 * 8 * 32) return;
  int lane = t & 31;
  int nt   = (t >> 5) & 7;
  int ks   = (t >> 8) & 3;
  int m    = t >> 10;
  const float* W = (m == 0) ? Wself
                 : (m <= R_) ? (Wword + (size_t)(m - 1) * D_ * D_)
                 : Wcross;
  int col   = nt * 16 + (lane & 15);
  int kbase = ks * 32 + ((lane < 16) ? 0 : 16);
  _Float16* dst = wf + (size_t)t * 16;
#pragma unroll
  for (int j = 0; j < 16; ++j)
    dst[j] = (_Float16)W[(size_t)(kbase + j) * D_ + col];
}

// ---------------------------------------------------------------------------
// LayerNorm (no affine) + cast to f16. One wave32 per row, 4 floats per lane.
// ---------------------------------------------------------------------------
__global__ void ln_kernel(const float* __restrict__ x, _Float16* __restrict__ xh, int N) {
  int row  = (blockIdx.x * blockDim.x + threadIdx.x) >> 5;
  int lane = threadIdx.x & 31;
  if (row >= N) return;
  const float* rp = x + (size_t)row * D_;
  float v[4];
  float s = 0.0f;
#pragma unroll
  for (int j = 0; j < 4; ++j) { v[j] = rp[lane * 4 + j]; s += v[j]; }
#pragma unroll
  for (int o = 16; o > 0; o >>= 1) s += __shfl_xor(s, o, 32);
  float mu = s * (1.0f / (float)D_);
  float q = 0.0f;
#pragma unroll
  for (int j = 0; j < 4; ++j) { v[j] -= mu; q += v[j] * v[j]; }
#pragma unroll
  for (int o = 16; o > 0; o >>= 1) q += __shfl_xor(q, o, 32);
  float rinv = rsqrtf(q * (1.0f / (float)D_) + 1e-5f);
  _Float16* dp = xh + (size_t)row * D_ + lane * 4;
#pragma unroll
  for (int j = 0; j < 4; ++j) dp[j] = (_Float16)(v[j] * rinv);
}

// ---------------------------------------------------------------------------
// C[N,128] = A_h[N,128] @ W (pre-packed fragments), f32 out.
// 256 threads = 8 waves; wave w owns rows [blk*128 + 16w, +16), all 128 cols.
// Per K-step: one clause of 16 b128 loads (8 B frags) + 2 b128 A loads, one
// wait, then 8 back-to-back v_wmma_f32_16x16x32_f16.
// ---------------------------------------------------------------------------
__global__ __launch_bounds__(256) void gemm_wmma_kernel(
    const _Float16* __restrict__ Ah, const _Float16* __restrict__ Wf,
    float* __restrict__ Cout, int N) {
  int lane    = threadIdx.x & 31;
  int wv      = threadIdx.x >> 5;
  int rowBase = blockIdx.x * 128 + wv * 16;
  int arow    = rowBase + (lane & 15);
  if (arow >= N) arow = N - 1;  // clamp; tail rows are never stored
  const _Float16* arp = Ah + (size_t)arow * D_;
  const _Float16* wfp = Wf + (size_t)lane * 16;

  v8f acc[8];
#pragma unroll
  for (int i = 0; i < 8; ++i)
#pragma unroll
    for (int j = 0; j < 8; ++j) acc[i][j] = 0.0f;

#pragma unroll
  for (int ks = 0; ks < 4; ++ks) {
    // A frag, contiguous k order (matches packed B): lane<16 -> M=lane, K = ks*32 + {0..15}
    //                  lane>=16 -> M=lane-16, K = ks*32 + {16..31}
    int kb = ks * 32 + ((lane < 16) ? 0 : 16);
    v8h lo = *(const v8h*)(arp + kb);
    v8h hi = *(const v8h*)(arp + kb + 8);
    v16h a;
#pragma unroll
    for (int j = 0; j < 8; ++j) { a[j] = lo[j]; a[j + 8] = hi[j]; }

    // Preload all 8 B fragments for this K-step (one 16-load clause).
    v16h b[8];
#pragma unroll
    for (int nt = 0; nt < 8; ++nt)
      b[nt] = *(const v16h*)(wfp + (size_t)(ks * 8 + nt) * 32 * 16);

#pragma unroll
    for (int nt = 0; nt < 8; ++nt)
      acc[nt] = __builtin_amdgcn_wmma_f32_16x16x32_f16(
          false, a, false, b[nt], (short)0, acc[nt], false, false);
  }

  // C/D layout: VGPR i -> M=i (lanes 0-15), M=8+i (lanes 16-31); N=lane%16.
  int rOff = (lane < 16) ? 0 : 8;
  int col0 = lane & 15;
#pragma unroll
  for (int nt = 0; nt < 8; ++nt)
#pragma unroll
    for (int i = 0; i < 8; ++i) {
      int row = rowBase + rOff + i;
      if (row < N) Cout[(size_t)row * D_ + nt * 16 + col0] = acc[nt][i];
    }
}

// ---------------------------------------------------------------------------
// Per-(node,head) source/target logits: al = <xw[n,h,:], a_s>, ar = <., a_d>.
// ---------------------------------------------------------------------------
__global__ void rel_logits_kernel(const float* __restrict__ xw,
                                  const float* __restrict__ a_s,
                                  const float* __restrict__ a_d,
                                  float* __restrict__ al, float* __restrict__ ar, int N) {
  int t = blockIdx.x * blockDim.x + threadIdx.x;
  if (t >= N * H_) return;
  int h = t & (H_ - 1);
  const float* xr = xw + (size_t)(t >> 3) * D_ + h * C_;
  float sa = 0.0f, sd = 0.0f;
#pragma unroll
  for (int c = 0; c < C_; ++c) {
    sa += xr[c] * a_s[h * C_ + c];
    sd += xr[c] * a_d[h * C_ + c];
  }
  al[t] = sa;
  ar[t] = sd;
}

__global__ void init_rel_kernel(unsigned* __restrict__ mkey, float* __restrict__ den,
                                float* __restrict__ num, int N) {
  int t = blockIdx.x * blockDim.x + threadIdx.x;
  if (t < N * H_) { mkey[t] = fkey(NEGV); den[t] = 0.0f; }
  if (t < N * D_) num[t] = 0.0f;
}

// Segment-max of leaky-relu logits over incoming edges (order-free atomicMax
// on an order-preserving uint mapping of the float).
__global__ void edge_max_kernel(const int* __restrict__ ei, const int* __restrict__ et,
                                const float* __restrict__ al, const float* __restrict__ ar,
                                unsigned* __restrict__ mkey, int E, int r) {
  int e = blockIdx.x * blockDim.x + threadIdx.x;
  if (e >= E || et[e] != r) return;
  int s = ei[e], d = ei[E + e];
#pragma unroll
  for (int h = 0; h < H_; ++h) {
    float l = lreluf(al[s * H_ + h] + ar[d * H_ + h]);
    atomicMax(&mkey[d * H_ + h], fkey(l));
  }
}

// w = exp(logit - m[dst]); den[dst,h] += w; num[dst,h,:] += w * xw[src,h,:].
__global__ void edge_acc_kernel(const int* __restrict__ ei, const int* __restrict__ et,
                                const float* __restrict__ al, const float* __restrict__ ar,
                                const unsigned* __restrict__ mkey, const float* __restrict__ xw,
                                float* __restrict__ den, float* __restrict__ num, int E, int r) {
  int e = blockIdx.x * blockDim.x + threadIdx.x;
  if (e >= E || et[e] != r) return;
  int s = ei[e], d = ei[E + e];
  __builtin_prefetch(xw + (size_t)s * D_, 0, 1);  // global_prefetch_b8
#pragma unroll
  for (int h = 0; h < H_; ++h) {
    float m = fmaxf(funkey(mkey[d * H_ + h]), NEGV);
    float l = lreluf(al[s * H_ + h] + ar[d * H_ + h]);
    float w = expf(l - m);
    atomicAdd(&den[d * H_ + h], w);
    const float* xs = xw + (size_t)s * D_ + h * C_;
    float* nd = num + (size_t)d * D_ + h * C_;
#pragma unroll
    for (int c = 0; c < C_; ++c) atomicAdd(&nd[c], w * xs[c]);
  }
}

// out = gelu(num/den + bias_word[r]) cast to f16 for the W_cross GEMM;
// also record per-node validity (any incoming edge of this relation).
__global__ void node_finish_kernel(const float* __restrict__ num, const float* __restrict__ den,
                                   const float* __restrict__ bias, _Float16* __restrict__ gh,
                                   float* __restrict__ validf, int N) {
  int t = blockIdx.x * blockDim.x + threadIdx.x;
  if (t >= N * H_) return;
  int n = t >> 3, h = t & (H_ - 1);
  float dn = den[t];
  float dd = dn > 0.0f ? dn : 1.0f;
  if (h == 0) validf[n] = (den[(size_t)n * H_] > 0.0f) ? 1.0f : 0.0f;
  const float* np = num + (size_t)n * D_ + h * C_;
  _Float16* gp = gh + (size_t)n * D_ + h * C_;
#pragma unroll
  for (int c = 0; c < C_; ++c) {
    float o = np[c] / dd + bias[h * C_ + c];
    gp[c] = (_Float16)geluf(o);
  }
}

// Star-graph language attention: softmax over the R+1 candidate rows per
// (node, head), weighted sum, + bias, gelu, + residual.
__global__ void lang_kernel(const float* __restrict__ feats, const float* __restrict__ validf,
                            const float* __restrict__ asl, const float* __restrict__ adl,
                            const float* __restrict__ bias, const float* __restrict__ xinp,
                            float* __restrict__ out, int N) {
  int t = blockIdx.x * blockDim.x + threadIdx.x;
  if (t >= N * H_) return;
  int n = t >> 3, h = t & (H_ - 1);
  size_t base = (size_t)n * D_ + h * C_;
  size_t slab = (size_t)N * D_;

  float v[R_ + 1][C_];
#pragma unroll
  for (int s = 0; s < R_ + 1; ++s) {
    const float* fp = feats + (size_t)s * slab + base;
#pragma unroll
    for (int c = 0; c < C_; ++c) v[s][c] = fp[c];
  }

  float arv = 0.0f;
#pragma unroll
  for (int c = 0; c < C_; ++c) arv += v[0][c] * adl[h * C_ + c];

  float lg[R_ + 1];
#pragma unroll
  for (int s = 0; s < R_ + 1; ++s) {
    float a = 0.0f;
#pragma unroll
    for (int c = 0; c < C_; ++c) a += v[s][c] * asl[h * C_ + c];
    float l = lreluf(a + arv);
    bool ok = (s == 0) || (validf[(size_t)(s - 1) * N + n] > 0.0f);
    lg[s] = ok ? l : NEGV;
  }
  float mx = lg[0];
#pragma unroll
  for (int s = 1; s < R_ + 1; ++s) mx = fmaxf(mx, lg[s]);
  float w[R_ + 1], sum = 0.0f;
#pragma unroll
  for (int s = 0; s < R_ + 1; ++s) { w[s] = expf(lg[s] - mx); sum += w[s]; }
  float inv = 1.0f / sum;

#pragma unroll
  for (int c = 0; c < C_; ++c) {
    float o = 0.0f;
#pragma unroll
    for (int s = 0; s < R_ + 1; ++s) o += w[s] * v[s][c];
    o = o * inv + bias[h * C_ + c];
    out[base + c] = geluf(o) + xinp[base + c];
  }
}

// ---------------------------------------------------------------------------
extern "C" void kernel_launch(void* const* d_in, const int* in_sizes, int n_in,
                              void* d_out, int out_size, void* d_ws, size_t ws_size,
                              hipStream_t stream) {
  const float* x_inp        = (const float*)d_in[0];
  // d_in[1] node_type: unused by the reference
  const int*   edge_index   = (const int*)d_in[2];
  const int*   edge_type    = (const int*)d_in[3];
  const float* W_self       = (const float*)d_in[4];
  const float* W_word       = (const float*)d_in[5];
  const float* att_src_word = (const float*)d_in[6];
  const float* att_dst_word = (const float*)d_in[7];
  const float* bias_word    = (const float*)d_in[8];
  const float* W_cross      = (const float*)d_in[9];
  const float* att_src_lang = (const float*)d_in[10];
  const float* att_dst_lang = (const float*)d_in[11];
  const float* bias_lang    = (const float*)d_in[12];

  const int N = in_sizes[1];  // node_type has N entries
  const int E = in_sizes[3];  // edge_type has E entries

  // Carve workspace (256-byte aligned slabs).
  char* w = (char*)d_ws;
  auto carve = [&](size_t bytes) {
    char* p = w;
    w += (bytes + 255) & ~(size_t)255;
    return (void*)p;
  };
  _Float16* wfrag  = (_Float16*)carve((size_t)7 * D_ * D_ * sizeof(_Float16));
  _Float16* xh     = (_Float16*)carve((size_t)N * D_ * sizeof(_Float16));
  _Float16* gh     = (_Float16*)carve((size_t)N * D_ * sizeof(_Float16));
  float*    xw     = (float*)carve((size_t)N * D_ * sizeof(float));
  float*    feats  = (float*)carve((size_t)(R_ + 1) * N * D_ * sizeof(float));
  float*    al     = (float*)carve((size_t)N * H_ * sizeof(float));
  float*    ar     = (float*)carve((size_t)N * H_ * sizeof(float));
  unsigned* mkey   = (unsigned*)carve((size_t)N * H_ * sizeof(unsigned));
  float*    den    = (float*)carve((size_t)N * H_ * sizeof(float));
  float*    validf = (float*)carve((size_t)R_ * N * sizeof(float));

  const int T = 256;
  dim3 blk(T);

  pack_w_kernel<<<(7 * 4 * 8 * 32 + T - 1) / T, blk, 0, stream>>>(W_self, W_word, W_cross, wfrag);
  ln_kernel<<<(N + 3) / 4, dim3(128), 0, stream>>>(x_inp, xh, N);

  dim3 gemm_grid((N + 127) / 128);
  // feats[0] = ln(x) @ W_self
  gemm_wmma_kernel<<<gemm_grid, blk, 0, stream>>>(xh, wfrag, feats, N);

  for (int r = 0; r < R_; ++r) {
    float* numr = feats + (size_t)(r + 1) * N * D_;
    const _Float16* wword_f = wfrag + (size_t)(1 + r) * D_ * D_;
    const _Float16* wcross_f = wfrag + (size_t)6 * D_ * D_;

    gemm_wmma_kernel<<<gemm_grid, blk, 0, stream>>>(xh, wword_f, xw, N);
    rel_logits_kernel<<<(N * H_ + T - 1) / T, blk, 0, stream>>>(
        xw, att_src_word + (size_t)r * H_ * C_, att_dst_word + (size_t)r * H_ * C_, al, ar, N);
    init_rel_kernel<<<(N * D_ + T - 1) / T, blk, 0, stream>>>(mkey, den, numr, N);
    edge_max_kernel<<<(E + T - 1) / T, blk, 0, stream>>>(edge_index, edge_type, al, ar, mkey, E, r);
    edge_acc_kernel<<<(E + T - 1) / T, blk, 0, stream>>>(edge_index, edge_type, al, ar, mkey,
                                                         xw, den, numr, E, r);
    node_finish_kernel<<<(N * H_ + T - 1) / T, blk, 0, stream>>>(
        numr, den, bias_word + (size_t)r * D_, gh, validf + (size_t)r * N, N);
    // feats[r+1] = gelu(out_r) @ W_cross (overwrites the num slab)
    gemm_wmma_kernel<<<gemm_grid, blk, 0, stream>>>(gh, wcross_f, numr, N);
  }

  lang_kernel<<<(N * H_ + T - 1) / T, blk, 0, stream>>>(
      feats, validf, att_src_lang, att_dst_lang, bias_lang, x_inp, (float*)d_out, N);
}
